// Model_90666759619142
// MI455X (gfx1250) — matrix operand order repaired and checked by reading, not compiled, not run on hardware
//
#include <hip/hip_runtime.h>
#include <stdint.h>

// ---------------------------------------------------------------------------
// SRU (4 layers) + vocab projection for MI455X (gfx1250, wave32, WMMA + TDM).
//  - GEMMs: v_wmma_f32_16x16x32_bf16; tiles DMA'd global->LDS by the Tensor
//    Data Mover (tensor_load_to_lds), double-buffered on s_wait_tensorcnt.
//  - Gate sigmoids fused into the GEMM epilogue; gate tensor stored bf16.
//  - Scan: B*H/2 threads, f32 cell state in registers, bf16 in/out.
// ---------------------------------------------------------------------------

typedef unsigned short u16t;
typedef __bf16 v16bf __attribute__((ext_vector_type(16)));
typedef float  v8f   __attribute__((ext_vector_type(8)));
typedef unsigned int v4u __attribute__((ext_vector_type(4)));
typedef int          v8i __attribute__((ext_vector_type(8)));
typedef int          v4i __attribute__((ext_vector_type(4)));

constexpr int SL = 512, SB = 32, SH = 1024, SDEPTH = 4, SV = 4096;
constexpr int M_ROWS  = SL * SB;   // 16384
constexpr int N_GATES = 3 * SH;    // 3072

__device__ __forceinline__ u16t f2bf(float f) {
  uint32_t u = __builtin_bit_cast(uint32_t, f);
  u += 0x7FFFu + ((u >> 16) & 1u);             // round-to-nearest-even
  return (u16t)(u >> 16);
}
__device__ __forceinline__ float bflo(uint32_t u) {
  return __builtin_bit_cast(float, u << 16);
}
__device__ __forceinline__ float bfhi(uint32_t u) {
  return __builtin_bit_cast(float, u & 0xFFFF0000u);
}
__device__ __forceinline__ uint32_t packbf(float a, float b) {
  return (uint32_t)f2bf(a) | ((uint32_t)f2bf(b) << 16);
}

// ---- Tensor Data Mover: 2D tile (global, row-major, elem=2B) -> LDS ------
// D# per CDNA5 ISA 8.3/8.4: group0 = {flags, lds_addr, global_addr, type=2},
// group1 = {data_size, tensor_dim0/1, tile_dim0/1, tensor_dim0_stride}.
__device__ __forceinline__ uint32_t lds_off(const void* p) {
  return (uint32_t)(uintptr_t)p;     // generic LDS ptr: addr[31:0] = LDS offset
}
__device__ __forceinline__ void tdm_load_2d(const void* gbase, uint32_t ldsa,
                                            uint32_t row_elems, uint32_t rows,
                                            uint32_t stride_elems,
                                            uint32_t tile_w, uint32_t tile_h) {
  const uint64_t ga = (uint64_t)(uintptr_t)gbase;
  v4u g0 = { 1u,                                   // count=1 (valid), user mode
             ldsa,                                 // lds_addr [63:32]
             (uint32_t)ga,                         // global_addr [95:64]
             (uint32_t)((ga >> 32) & 0x01FFFFFFu)  // global_addr [120:96]
                 | (2u << 30) };                   // type=2 ("image") [127:126]
  v8i g1 = { (int)(1u << 16),                                    // data_size=1 (2B)
             (int)((row_elems & 0xFFFFu) << 16),                 // tensor_dim0 lo
             (int)((row_elems >> 16) | ((rows & 0xFFFFu) << 16)),// dim0 hi | dim1 lo
             (int)((rows >> 16) | (tile_w << 16)),               // dim1 hi | tile_dim0
             (int)tile_h,                                        // tile_dim1 (dim2=0)
             (int)stride_elems,                                  // dim0_stride lo
             0, 0 };
  v4i z4 = {0, 0, 0, 0};
#if __clang_major__ >= 23
  v8i z8 = {0, 0, 0, 0, 0, 0, 0, 0};
  __builtin_amdgcn_tensor_load_to_lds(g0, g1, z4, z4, z8, 0);
#else
  __builtin_amdgcn_tensor_load_to_lds(g0, g1, z4, z4, 0);
#endif
}

// ---- f32 -> bf16 flat ----------------------------------------------------
__global__ __launch_bounds__(256) void cvt_f32_bf16(const float* __restrict__ src,
                                                    u16t* __restrict__ dst, int n) {
  int i = blockIdx.x * 256 + threadIdx.x;
  if (i < n) dst[i] = f2bf(src[i]);
}

// ---- W [DEPTH][K=SH][N=3SH] -> bf16 transposed [DEPTH][N][K] -------------
__global__ __launch_bounds__(256) void cvt_w_T(const float* __restrict__ w,
                                               u16t* __restrict__ wt, int n) {
  int i = blockIdx.x * 256 + threadIdx.x;
  if (i < n) {
    int l   = i / (SH * N_GATES);
    int rem = i % (SH * N_GATES);
    int k   = rem / N_GATES;
    int nn  = rem % N_GATES;
    wt[(size_t)l * SH * N_GATES + (size_t)nn * SH + k] = f2bf(w[i]);
  }
}

// ---- bf16 GEMM: C[M,N] = A[M,K] * BT[N,K]^T ------------------------------
// MODE 0: C=f32, +bias[n]            (vocab projection)
// MODE 1: C=bf16, SRU gate epilogue  (z raw; f,r -> sigmoid(+bias[2H]))
struct Frag32B { uint4 lo, hi; };

template <int MODE, int NDIM>
__global__ __launch_bounds__(128) void gemm_bf16_wmma(
    const u16t* __restrict__ A,   // [M][K]
    const u16t* __restrict__ BT,  // [N][K]
    void* __restrict__ Cv,
    const float* __restrict__ bias,
    int M)
{
  constexpr int K = SH;
  constexpr int BM = 64, BN = 64, BK = 64;
  constexpr int NIT = K / BK;                 // 16
  __shared__ __align__(16) u16t As[2][BM][BK];
  __shared__ __align__(16) u16t Bs[2][BN][BK];

  const int tid  = threadIdx.x;
  const int wave = tid >> 5;
  const int lane = tid & 31;
  const int lm   = lane & 15;      // M (A) / N (B) index inside fragment
  const int lg   = lane >> 4;      // lane half -> K interleave group

  const int bm = blockIdx.y * BM;
  const int bn = blockIdx.x * BN;
  const int wm = (wave >> 1) * 32;
  const int wn = (wave & 1) * 32;

  v8f acc[2][2] = {};

  // prologue: DMA first tile pair into buffer 0 (wave 0 drives the TDM)
  if (wave == 0) {
    tdm_load_2d(A  + (size_t)bm * K, lds_off(&As[0][0][0]), K, (uint32_t)M, K, BK, BM);
    tdm_load_2d(BT + (size_t)bn * K, lds_off(&Bs[0][0][0]), K, (uint32_t)NDIM, K, BK, BN);
  }

  for (int it = 0; it < NIT; ++it) {
    const int cur = it & 1;
    if (wave == 0) {
      if (it + 1 < NIT) {   // issue next pair, then wait for current pair
        tdm_load_2d(A  + (size_t)bm * K + (it + 1) * BK,
                    lds_off(&As[cur ^ 1][0][0]), K, (uint32_t)M, K, BK, BM);
        tdm_load_2d(BT + (size_t)bn * K + (it + 1) * BK,
                    lds_off(&Bs[cur ^ 1][0][0]), K, (uint32_t)NDIM, K, BK, BN);
        __builtin_amdgcn_s_wait_tensorcnt(2);   // in-order: current pair done
      } else {
        __builtin_amdgcn_s_wait_tensorcnt(0);
      }
    }
    __syncthreads();        // release all waves to consume buffer `cur`

#pragma unroll
    for (int ks = 0; ks < BK; ks += 32) {
      v16bf afr[2], bfr[2];
#pragma unroll
      for (int i = 0; i < 2; ++i) {   // A frag: ISA K-interleave across lane halves
        const uint4* ap = reinterpret_cast<const uint4*>(&As[cur][wm + i * 16 + lm][0]);
        Frag32B f{ap[(ks >> 3) + lg], ap[(ks >> 3) + 2 + lg]};
        afr[i] = __builtin_bit_cast(v16bf, f);
      }
#pragma unroll
      for (int i = 0; i < 2; ++i) {   // B frag: ISA K-interleave across lane halves
        const uint4* bp = reinterpret_cast<const uint4*>(&Bs[cur][wn + i * 16 + lm][0]);
        Frag32B f{bp[(ks >> 3) + lg], bp[(ks >> 3) + 2 + lg]};
        bfr[i] = __builtin_bit_cast(v16bf, f);
      }
#pragma unroll
      for (int i = 0; i < 2; ++i)
#pragma unroll
        for (int j = 0; j < 2; ++j)
          acc[i][j] = __builtin_amdgcn_wmma_f32_16x16x32_bf16(
              false, afr[i], false, bfr[j], (short)0, acc[i][j], false, false);
    }
    __syncthreads();        // all reads of buffer `cur` done; it may be refilled
  }

  // epilogue: D layout -> row = lg*8 + r, col = lm  (bias hoisted per column)
#pragma unroll
  for (int j = 0; j < 2; ++j) {
    const int n = bn + wn + j * 16 + lm;
    float bb;
    if (MODE == 0) bb = bias[n];
    else           bb = (n >= SH) ? bias[n - SH] : 0.f;
#pragma unroll
    for (int i = 0; i < 2; ++i)
#pragma unroll
      for (int r = 0; r < 8; ++r) {
        const int m = bm + wm + i * 16 + lg * 8 + r;
        float v = acc[i][j][r];
        if (MODE == 0) {
          reinterpret_cast<float*>(Cv)[(size_t)m * NDIM + n] = v + bb;
        } else {
          if (n >= SH)  // f / r gates: sigmoid
            v = 1.f / (1.f + __expf(-(v + bb)));
          reinterpret_cast<u16t*>(Cv)[(size_t)m * NDIM + n] = f2bf(v);
        }
      }
  }
}

// ---- SRU recurrence + highway: one thread per (b, h-pair) ----------------
// G: bf16 [L*B][3H] with f,r already sigmoided. Xin/Hout: bf16 [L*B][H].
__global__ __launch_bounds__(256) void sru_scan(
    const u16t* __restrict__ G, const u16t* __restrict__ Xin,
    const float* __restrict__ c0, u16t* __restrict__ Hout,
    float* __restrict__ cLast)
{
  const int idx = blockIdx.x * 256 + threadIdx.x;   // 0 .. B*H/2-1
  const int b  = idx >> 9;                          // / (H/2)
  const int hp = (idx & 511) * 2;                   // even h
  float2 c = *reinterpret_cast<const float2*>(&c0[b * SH + hp]);

  for (int t = 0; t < SL; ++t) {
    const size_t row = (size_t)(t * SB + b);
    const u16t* g = &G[row * N_GATES];
    const uint32_t zu = *reinterpret_cast<const uint32_t*>(&g[hp]);
    const uint32_t fu = *reinterpret_cast<const uint32_t*>(&g[SH + hp]);
    const uint32_t ru = *reinterpret_cast<const uint32_t*>(&g[2 * SH + hp]);
    const uint32_t xu = *reinterpret_cast<const uint32_t*>(&Xin[row * SH + hp]);
    const float f0 = bflo(fu), f1 = bfhi(fu);
    const float r0 = bflo(ru), r1 = bfhi(ru);
    c.x = f0 * c.x + (1.f - f0) * bflo(zu);
    c.y = f1 * c.y + (1.f - f1) * bfhi(zu);
    const float h0 = r0 * c.x + (1.f - r0) * bflo(xu);
    const float h1 = r1 * c.y + (1.f - r1) * bfhi(xu);
    *reinterpret_cast<uint32_t*>(&Hout[row * SH + hp]) = packbf(h0, h1);
  }
  *reinterpret_cast<float2*>(&cLast[b * SH + hp]) = c;
}

// ---------------------------------------------------------------------------
extern "C" void kernel_launch(void* const* d_in, const int* in_sizes, int n_in,
                              void* d_out, int out_size, void* d_ws, size_t ws_size,
                              hipStream_t stream) {
  (void)in_sizes; (void)n_in; (void)out_size; (void)ws_size;
  const float* x      = (const float*)d_in[0];
  const float* hidden = (const float*)d_in[1];
  // d_in[2] = lens (int64) : unused by the reference computation
  const float* W      = (const float*)d_in[3];
  const float* bgate  = (const float*)d_in[4];
  const float* Wout   = (const float*)d_in[5];
  const float* bout   = (const float*)d_in[6];

  float* logits     = (float*)d_out;
  float* hidden_out = logits + (size_t)M_ROWS * SV;

  char* ws = (char*)d_ws;
  auto carve = [&](size_t bytes) -> char* {
    char* p = ws; ws += (bytes + 255) & ~(size_t)255; return p;
  };
  u16t* Xb0 = (u16t*)carve((size_t)M_ROWS * SH * 2);            // layer io ping
  u16t* Xb1 = (u16t*)carve((size_t)M_ROWS * SH * 2);            // layer io pong
  u16t* WbT = (u16t*)carve((size_t)SDEPTH * SH * N_GATES * 2);  // W^T bf16
  u16t* WoB = (u16t*)carve((size_t)SV * SH * 2);                // W_out bf16 ([N][K] already)
  u16t* G   = (u16t*)carve((size_t)M_ROWS * N_GATES * 2);       // gates bf16

  const int nXH   = M_ROWS * SH;             // 16,777,216
  const int nWall = SDEPTH * SH * N_GATES;   // 12,582,912
  const int nWo   = SV * SH;                 // 4,194,304
  dim3 blk(256);

  cvt_f32_bf16<<<dim3((nXH + 255) / 256), blk, 0, stream>>>(x, Xb0, nXH);
  cvt_w_T     <<<dim3((nWall + 255) / 256), blk, 0, stream>>>(W, WbT, nWall);
  cvt_f32_bf16<<<dim3((nWo + 255) / 256), blk, 0, stream>>>(Wout, WoB, nWo);

  dim3 gGemmU(N_GATES / 64, M_ROWS / 64);
  dim3 gGemmO(SV / 64, M_ROWS / 64);
  dim3 gScan((SB * SH / 2) / 256);

  u16t* cur = Xb0;
  u16t* nxt = Xb1;
  for (int l = 0; l < SDEPTH; ++l) {
    gemm_bf16_wmma<1, N_GATES><<<gGemmU, dim3(128), 0, stream>>>(
        cur, WbT + (size_t)l * SH * N_GATES, (void*)G,
        bgate + (size_t)l * 2 * SH, M_ROWS);
    sru_scan<<<gScan, blk, 0, stream>>>(
        G, cur, hidden + (size_t)l * SB * SH, nxt,
        hidden_out + (size_t)l * SB * SH);
    u16t* tmp = cur; cur = nxt; nxt = tmp;
  }

  gemm_bf16_wmma<0, SV><<<gGemmO, dim3(128), 0, stream>>>(
      cur, WoB, (void*)logits, bout, M_ROWS);
}
